// CausalAttention_4097398800518
// MI455X (gfx1250) — matrix operand order repaired and checked by reading, not compiled, not run on hardware
//
#include <hip/hip_runtime.h>

// ---------------------------------------------------------------------------
// CausalAttention for MI455X (gfx1250, wave32, WMMA bf16 16x16x32, f32 accum)
//
// B=16, CQ=CK=CH=512, NH=8, H=W=32 -> S=1024, DH=64
//
// ws layout (bf16 elements):
//   Wq [512*512], Wk, Wv                      (weight-norm'd, [Cout][Cin])
//   Xq [16*1024*512], Xk                      (inputs transposed to [B,S,C])
//   Qm [B,NH,S,DH] (pre-scaled by 1/8), Km [B,NH,S,DH], Vt [B,NH,DH,S]
// total ~85.5 MB
// ---------------------------------------------------------------------------

typedef __attribute__((ext_vector_type(16))) __bf16 v16bf;
typedef __attribute__((ext_vector_type(8)))  float  v8f;

#define NB   16
#define SQ   1024
#define CIN  512
#define CHN  512
#define NHD  8
#define DH   64

union Frag16 { v16bf v; uint4 u[2]; };

static __device__ inline v16bf ldfrag(const __bf16* p0, const __bf16* p1) {
  Frag16 f;
  f.u[0] = *(const uint4*)p0;
  f.u[1] = *(const uint4*)p1;
  return f.v;
}

static __device__ inline v8f wmma_bf16(v16bf a, v16bf b, v8f c) {
  return __builtin_amdgcn_wmma_f32_16x16x32_bf16(false, a, false, b, (short)0, c,
                                                 false, false);
}

// reduce across the 16 lanes of a half-wave (xor masks < 16 stay in-half)
static __device__ inline float rowmax16(float v) {
  v = fmaxf(v, __shfl_xor(v, 1, 32));
  v = fmaxf(v, __shfl_xor(v, 2, 32));
  v = fmaxf(v, __shfl_xor(v, 4, 32));
  v = fmaxf(v, __shfl_xor(v, 8, 32));
  return v;
}
static __device__ inline float rowsum16(float v) {
  v += __shfl_xor(v, 1, 32);
  v += __shfl_xor(v, 2, 32);
  v += __shfl_xor(v, 4, 32);
  v += __shfl_xor(v, 8, 32);
  return v;
}

// ---------------------------------------------------------------------------
// Kernel 1: weight-norm -> bf16 W, kept in [Cout][Cin] row-major
// grid: (512 rows, 3 mats), block 256
// ---------------------------------------------------------------------------
__global__ void wn_prep(const float* __restrict__ vq, const float* __restrict__ gq,
                        const float* __restrict__ vk, const float* __restrict__ gk,
                        const float* __restrict__ vv, const float* __restrict__ gv,
                        __bf16* __restrict__ Wq, __bf16* __restrict__ Wk,
                        __bf16* __restrict__ Wv) {
  const int row = blockIdx.x;
  const int mat = blockIdx.y;
  const int t   = threadIdx.x;
  const float* v = (mat == 0) ? vq : (mat == 1) ? vk : vv;
  const float* g = (mat == 0) ? gq : (mat == 1) ? gk : gv;
  __bf16*      W = (mat == 0) ? Wq : (mat == 1) ? Wk : Wv;

  const float* vr = v + (size_t)row * CIN;
  float a0 = vr[t], a1 = vr[t + 256];

  __shared__ float red[256];
  red[t] = a0 * a0 + a1 * a1;
  __syncthreads();
  for (int s = 128; s > 0; s >>= 1) {
    if (t < s) red[t] += red[t + s];
    __syncthreads();
  }
  const float scale = g[row] * rsqrtf(red[0]);
  __bf16* wr = W + (size_t)row * CIN;
  wr[t]       = (__bf16)(a0 * scale);
  wr[t + 256] = (__bf16)(a1 * scale);
}

// ---------------------------------------------------------------------------
// Kernel 2: [B,C,S] f32 -> [B,S,C] bf16 via 32x32 LDS tile transpose
// grid: (S/32=32, C/32=16, B*2=32), block 256 (32x8)
// ---------------------------------------------------------------------------
__global__ void convert_x(const float* __restrict__ query,
                          const float* __restrict__ key,
                          __bf16* __restrict__ Xq, __bf16* __restrict__ Xk) {
  const int which = blockIdx.z & 1;
  const int b     = blockIdx.z >> 1;
  const int stile = blockIdx.x;  // along S
  const int ctile = blockIdx.y;  // along C
  const int tx = threadIdx.x & 31, ty = threadIdx.x >> 5;

  const float* src = which ? key : query;
  __bf16*      dst = which ? Xk  : Xq;

  __shared__ float tile[32][33];
  const float* in = src + ((size_t)b * CIN + ctile * 32) * SQ + stile * 32;
#pragma unroll
  for (int yy = ty; yy < 32; yy += 8)
    tile[yy][tx] = in[(size_t)yy * SQ + tx];
  __syncthreads();
  __bf16* out = dst + ((size_t)b * SQ + stile * 32) * CIN + ctile * 32;
#pragma unroll
  for (int yy = ty; yy < 32; yy += 8)
    out[(size_t)yy * CIN + tx] = (__bf16)tile[tx][yy];
}

// ---------------------------------------------------------------------------
// Kernel 3: projection GEMM  Y = X(1024x512) @ W^T + bias  (bf16 WMMA)
// grid: (16 Mtiles * 8 Ntiles = 128, B=16, mat=3), block 256 = 8 waves
// wave -> 16x32 output (two 16x16 WMMA subtiles), K loop 512/32
// mat 0: Q [B,NH,S,DH] scaled 0.125 ; mat 1: K [B,NH,S,DH] ; mat 2: V^T [B,NH,DH,S]
// ---------------------------------------------------------------------------
__global__ __launch_bounds__(256) void proj_gemm(
    const __bf16* __restrict__ Xq, const __bf16* __restrict__ Xk,
    const __bf16* __restrict__ Wq, const __bf16* __restrict__ Wk,
    const __bf16* __restrict__ Wv, const float* __restrict__ bq,
    const float* __restrict__ bk, const float* __restrict__ bv,
    __bf16* __restrict__ Qm, __bf16* __restrict__ Km, __bf16* __restrict__ Vt) {
  const int mat   = blockIdx.z;
  const int b     = blockIdx.y;
  const int mtile = blockIdx.x & 15;
  const int ntile = blockIdx.x >> 4;
  const int w = threadIdx.x >> 5, lane = threadIdx.x & 31;
  const int lane15  = lane & 15;
  const int halfoff = (lane >= 16) ? 8 : 0;
  const int koffA   = (lane >= 16) ? 16 : 0;  // A-frag: K 0..15 / 16..31 per half
  const int koffB   = (lane >= 16) ? 16 : 0;  // B-frag: K 0..15 / 16..31 per half

  const int r0 = (w & 3) * 16;              // M sub-offset within 64 tile
  const int n0 = ntile * 64 + (w >> 2) * 32;  // N base (two subtiles)

  const __bf16* X    = (mat == 0) ? Xq : Xk;
  const __bf16* W    = (mat == 0) ? Wq : (mat == 1) ? Wk : Wv;
  const float*  bias = (mat == 0) ? bq : (mat == 1) ? bk : bv;

  const __bf16* arow  = X + ((size_t)b * SQ + mtile * 64 + r0 + lane15) * CIN;
  const __bf16* brow0 = W + (size_t)(n0 + lane15) * CIN + koffB;
  const __bf16* brow1 = brow0 + (size_t)16 * CIN;

  v8f c0 = {}, c1 = {};
#pragma unroll 4
  for (int k0 = 0; k0 < CIN; k0 += 32) {
    v16bf a  = ldfrag(arow + k0 + koffA, arow + k0 + 8 + koffA);
    v16bf b0 = ldfrag(brow0 + k0, brow0 + k0 + 8);
    v16bf b1 = ldfrag(brow1 + k0, brow1 + k0 + 8);
    c0 = wmma_bf16(a, b0, c0);
    c1 = wmma_bf16(a, b1, c1);
  }

#pragma unroll
  for (int sub = 0; sub < 2; ++sub) {
    v8f c = sub ? c1 : c0;
    const int col = n0 + sub * 16 + lane15;
    const float bsv = bias[col];
    const int h = col >> 6, d = col & 63;
#pragma unroll
    for (int r = 0; r < 8; ++r) {
      const int s = mtile * 64 + r0 + halfoff + r;
      float val = c[r] + bsv;
      if (mat == 0) {
        Qm[(((size_t)b * NHD + h) * SQ + s) * DH + d] = (__bf16)(val * 0.125f);
      } else if (mat == 1) {
        Km[(((size_t)b * NHD + h) * SQ + s) * DH + d] = (__bf16)val;
      } else {
        Vt[(((size_t)b * NHD + h) * DH + d) * SQ + s] = (__bf16)val;
      }
    }
  }
}

// ---------------------------------------------------------------------------
// Kernel 4: flash-style causal attention (strict lower-triangular, j < i)
// grid: (S/128 = 8, B*NH = 128), block 256 = 8 waves; wave owns 16 query rows
// P (f32 C-frag) -> bf16 A-frag re-layout via wave-private LDS slab.
// ---------------------------------------------------------------------------
__global__ __launch_bounds__(256) void attn_kernel(
    const __bf16* __restrict__ Qm, const __bf16* __restrict__ Km,
    const __bf16* __restrict__ Vt, float* __restrict__ out) {
  const int bh = blockIdx.y;  // b*NH + h
  const int qt = blockIdx.x;
  const int w = threadIdx.x >> 5, lane = threadIdx.x & 31;
  const int q0      = qt * 128 + w * 16;
  const int lane15  = lane & 15;
  const int halfoff = (lane >= 16) ? 8 : 0;
  const int koffA   = (lane >= 16) ? 16 : 0;
  const int koffB   = (lane >= 16) ? 16 : 0;

  __shared__ __align__(16) __bf16 Pbuf[8][16][72];  // 72-half row pad (bank)
  __bf16(*P)[72] = Pbuf[w];

  // Q A-fragments for this wave's 16 rows (DH=64 -> two K=32 steps)
  const __bf16* qrow = Qm + ((size_t)bh * SQ + q0 + lane15) * DH;
  const v16bf qa0 = ldfrag(qrow + koffA,      qrow + 8 + koffA);
  const v16bf qa1 = ldfrag(qrow + 32 + koffA, qrow + 40 + koffA);

  v8f o0 = {}, o1 = {}, o2 = {}, o3 = {};
  float m8[8], l8[8];
#pragma unroll
  for (int r = 0; r < 8; ++r) { m8[r] = -1e30f; l8[r] = 0.f; }

  const __bf16* Kbase = Km + (size_t)bh * SQ * DH;
  const __bf16* Vbase = Vt + (size_t)bh * DH * SQ;

  for (int jb = 0; jb < q0 + 16; jb += 64) {
    // ---- S = Q K^T for four 16-key subtiles -----------------------------
    v8f st[4];
#pragma unroll
    for (int jj = 0; jj < 4; ++jj) {
      const __bf16* krow = Kbase + (size_t)(jb + jj * 16 + lane15) * DH + koffB;
      v16bf kb0 = ldfrag(krow, krow + 8);        // d = 0..31
      v16bf kb1 = ldfrag(krow + 32, krow + 40);  // d = 32..63
      v8f acc = {};
      acc = wmma_bf16(qa0, kb0, acc);
      acc = wmma_bf16(qa1, kb1, acc);
      st[jj] = acc;
    }
    // ---- strict causal mask (attend only to j < i) ----------------------
#pragma unroll
    for (int jj = 0; jj < 4; ++jj) {
      const int j = jb + jj * 16 + lane15;
#pragma unroll
      for (int r = 0; r < 8; ++r) {
        const int i = q0 + halfoff + r;
        st[jj][r] = (j < i) ? st[jj][r] : -1e30f;
      }
    }
    // ---- online softmax -------------------------------------------------
    float mnew[8], alpha[8];
#pragma unroll
    for (int r = 0; r < 8; ++r) {
      float mx = fmaxf(fmaxf(st[0][r], st[1][r]), fmaxf(st[2][r], st[3][r]));
      mnew[r]  = fmaxf(m8[r], rowmax16(mx));
      alpha[r] = __expf(m8[r] - mnew[r]);
      m8[r]    = mnew[r];
    }
#pragma unroll
    for (int jj = 0; jj < 4; ++jj) {
      const int j = jb + jj * 16 + lane15;
#pragma unroll
      for (int r = 0; r < 8; ++r) {
        const int i = q0 + halfoff + r;
        const float p = (j < i) ? __expf(st[jj][r] - mnew[r]) : 0.f;
        st[jj][r] = p;
        P[halfoff + r][jj * 16 + lane15] = (__bf16)p;  // wave-private LDS
      }
    }
#pragma unroll
    for (int r = 0; r < 8; ++r) {
      float sm = rowsum16(st[0][r] + st[1][r] + st[2][r] + st[3][r]);
      l8[r] = alpha[r] * l8[r] + sm;
      o0[r] *= alpha[r]; o1[r] *= alpha[r];
      o2[r] *= alpha[r]; o3[r] *= alpha[r];
    }
    // drain this wave's DS stores before cross-lane LDS reads (CDNA5 DScnt)
    asm volatile("s_wait_dscnt 0" ::: "memory");

    // ---- O += P V  (re-layout P as A-fragments from LDS) ----------------
    const __bf16* prow = &P[lane15][0];
    v16bf pa0 = ldfrag(prow + koffA,      prow + 8 + koffA);  // keys jb+0..31
    v16bf pa1 = ldfrag(prow + 32 + koffA, prow + 40 + koffA);  // keys jb+32..63
#pragma unroll
    for (int dsub = 0; dsub < 4; ++dsub) {
      const __bf16* vrow = Vbase + (size_t)(dsub * 16 + lane15) * SQ + jb + koffB;
      v16bf vb0 = ldfrag(vrow, vrow + 8);
      v16bf vb1 = ldfrag(vrow + 32, vrow + 40);
      v8f& o = (dsub == 0) ? o0 : (dsub == 1) ? o1 : (dsub == 2) ? o2 : o3;
      o = wmma_bf16(pa0, vb0, o);
      o = wmma_bf16(pa1, vb1, o);
    }
  }

  // ---- normalize + store [B, CH, S]; lsum==0 guard zeroes row 0 ---------
  const int bI = bh >> 3, h = bh & 7;
#pragma unroll
  for (int dsub = 0; dsub < 4; ++dsub) {
    v8f o = (dsub == 0) ? o0 : (dsub == 1) ? o1 : (dsub == 2) ? o2 : o3;
    const int c = h * DH + dsub * 16 + lane15;
#pragma unroll
    for (int r = 0; r < 8; ++r) {
      const int s = q0 + halfoff + r;
      const float denom = l8[r];
      out[((size_t)bI * CHN + c) * SQ + s] = (denom > 0.f) ? (o[r] / denom) : 0.f;
    }
  }
}

// ---------------------------------------------------------------------------
extern "C" void kernel_launch(void* const* d_in, const int* in_sizes, int n_in,
                              void* d_out, int out_size, void* d_ws,
                              size_t ws_size, hipStream_t stream) {
  const float* query = (const float*)d_in[0];
  const float* key   = (const float*)d_in[1];
  const float* vq = (const float*)d_in[2];
  const float* gq = (const float*)d_in[3];
  const float* bq = (const float*)d_in[4];
  const float* vk = (const float*)d_in[5];
  const float* gk = (const float*)d_in[6];
  const float* bk = (const float*)d_in[7];
  const float* vv = (const float*)d_in[8];
  const float* gv = (const float*)d_in[9];
  const float* bv = (const float*)d_in[10];
  float* out = (float*)d_out;

  // workspace carve-up (bf16 elements); ~85.5 MB total
  __bf16* ws = (__bf16*)d_ws;
  constexpr size_t WSZ = (size_t)CHN * CIN;            // 262144
  constexpr size_t XSZ = (size_t)NB * SQ * CIN;        // 8388608
  constexpr size_t QSZ = (size_t)NB * NHD * SQ * DH;   // 8388608
  __bf16* Wq = ws;
  __bf16* Wk = Wq + WSZ;
  __bf16* Wv = Wk + WSZ;
  __bf16* Xq = Wv + WSZ;
  __bf16* Xk = Xq + XSZ;
  __bf16* Qm = Xk + XSZ;
  __bf16* Km = Qm + QSZ;
  __bf16* Vt = Km + QSZ;

  wn_prep<<<dim3(CHN, 3), 256, 0, stream>>>(vq, gq, vk, gk, vv, gv, Wq, Wk, Wv);
  convert_x<<<dim3(SQ / 32, CIN / 32, NB * 2), 256, 0, stream>>>(query, key, Xq, Xk);
  proj_gemm<<<dim3(128, NB, 3), 256, 0, stream>>>(Xq, Xk, Wq, Wk, Wv, bq, bk, bv,
                                                  Qm, Km, Vt);
  attn_kernel<<<dim3(SQ / 128, NB * NHD), 256, 0, stream>>>(Qm, Km, Vt, out);
}
